// GCRN_62105227100495
// MI455X (gfx1250) — matrix operand order repaired and checked by reading, not compiled, not run on hardware
//
#include <hip/hip_runtime.h>
#include <hip/hip_bf16.h>

typedef __attribute__((ext_vector_type(16))) _Float16 v16h;
typedef __attribute__((ext_vector_type(8)))  _Float16 v8h;
typedef __attribute__((ext_vector_type(8)))  float    v8f;

union V16 { v16h v; v8h h8[2]; _Float16 e[16]; };

#define WMMA_F32_F16(a, b, c) \
  __builtin_amdgcn_wmma_f32_16x16x32_f16(false, (a), false, (b), (short)0, (c), false, false)

constexpr int   B_ = 16, N_ = 1024, F_ = 64, G_ = 64, E_ = 3;
constexpr int   H1_ = 256, H2_ = 256, EMB_ = 128;
constexpr int   CH_ = E_ * G_;           // 192
constexpr float NEG_SLOPE = 0.2f;
constexpr float MASK_VAL  = -9.0e15f;

// ---------------------------------------------------------------------------
// K0: u1[e,f] = Wq[e,f,:].a1[e,:];  u2[e,f] = Wv[e,f,:].a2[e,:]
// (collapses Q/V projections: s1 = X@u1, s2 = X@u2)
// ---------------------------------------------------------------------------
__global__ void k_uvec(const float* __restrict__ Wq, const float* __restrict__ Wv,
                       const float* __restrict__ a1, const float* __restrict__ a2,
                       float* __restrict__ u1, float* __restrict__ u2) {
  int t = threadIdx.x;
  if (t >= E_ * F_) return;
  int e = t / F_, f = t % F_;
  float r1 = 0.f, r2 = 0.f;
#pragma unroll 4
  for (int g = 0; g < G_; ++g) {
    r1 += Wq[(e * F_ + f) * G_ + g] * a1[e * G_ + g];
    r2 += Wv[(e * F_ + f) * G_ + g] * a2[e * G_ + g];
  }
  u1[t] = r1;
  u2[t] = r2;
}

// ---------------------------------------------------------------------------
// Kswz: convert a row-major f32 weight [K x Nout] (batched) into f16
// B-fragment order: frag(mat,kc,nt)[lane][idx] = W[kc*32 + (idx&7) + 8*(lane>>4) + 16*(idx>>3)]
//                                                 [nt*16 + (lane&15)]
// Consumers then load one whole fragment per lane as a 32-byte vector.
// ---------------------------------------------------------------------------
__global__ void k_wswz(const float* __restrict__ W, _Float16* __restrict__ Wz,
                       int K, int Nout, int nmat) {
  const int kcN = K / 32, ntN = Nout / 16;
  const int total = nmat * kcN * ntN * 32;
  int t = blockIdx.x * blockDim.x + threadIdx.x;
  if (t >= total) return;
  int lane = t & 31;
  int rest = t >> 5;
  int nt = rest % ntN; rest /= ntN;
  int kc = rest % kcN;
  int mat = rest / kcN;
  int n = lane & 15, hi = lane >> 4;
  const float* Wm = W + (size_t)mat * K * Nout;
  _Float16* dst = Wz + (size_t)t * 16;
#pragma unroll
  for (int idx = 0; idx < 16; ++idx) {
    int k = kc * 32 + (idx & 7) + 8 * hi + 16 * (idx >> 3);
    dst[idx] = (_Float16)Wm[(size_t)k * Nout + nt * 16 + n];
  }
}

// ---------------------------------------------------------------------------
// K1: per (b,e,16-row tile): Wh = X @ Ws[e] via WMMA, stored into the
// swizzled B-fragment layout Whs that k_attn consumes; plus s1/s2 mat-vecs.
// ---------------------------------------------------------------------------
__global__ void k_proj(const float* __restrict__ X, const _Float16* __restrict__ Wsz,
                       const float* __restrict__ u1, const float* __restrict__ u2,
                       _Float16* __restrict__ Whs, float* __restrict__ s1,
                       float* __restrict__ s2) {
  const int lane = threadIdx.x & 31;
  const int wave = threadIdx.x >> 5;
  const int tile = blockIdx.x * 4 + wave;        // 3072 tiles
  const int i0   = (tile & 63) * 16;
  const int e    = (tile >> 6) % E_;
  const int b    = tile / (64 * E_);
  const int m    = lane & 15;
  const int hi   = lane >> 4;
  const int be   = b * E_ + e;

  const float* Xt = X + ((size_t)(b * N_) + i0) * F_;

  // A fragments of X (16x32 each): two contiguous float4-pairs per half
  V16 xa[2];
#pragma unroll
  for (int t = 0; t < 2; ++t) {
    const float* base = Xt + m * F_ + t * 32 + 8 * hi;
    float4 q0 = *reinterpret_cast<const float4*>(base);
    float4 q1 = *reinterpret_cast<const float4*>(base + 4);
    float4 q2 = *reinterpret_cast<const float4*>(base + 16);
    float4 q3 = *reinterpret_cast<const float4*>(base + 20);
    float qs[16] = {q0.x, q0.y, q0.z, q0.w, q1.x, q1.y, q1.z, q1.w,
                    q2.x, q2.y, q2.z, q2.w, q3.x, q3.y, q3.z, q3.w};
#pragma unroll
    for (int idx = 0; idx < 16; ++idx) xa[t].e[idx] = (_Float16)qs[idx];
  }

  const int chunk = i0 >> 5;       // 32-row K-chunk index for k_attn
  const int wb    = i0 & 16;       // this tile's offset inside the chunk

#pragma unroll
  for (int nt = 0; nt < 4; ++nt) {
    v8f c = {};
#pragma unroll
    for (int t = 0; t < 2; ++t) {
      v16h bw = *reinterpret_cast<const v16h*>(
          Wsz + ((size_t)((e * 2 + t) * 4 + nt) * 32 + lane) * 16);
      c = WMMA_F32_F16(xa[t].v, bw, c);
    }
    // scatter into Whs fragment layout
#pragma unroll
    for (int r = 0; r < 8; ++r) {
      int jin = wb + r + 8 * hi;           // row within 32-chunk
      int hic = (jin >> 3) & 1, idq = (jin & 7) + 8 * (jin >> 4);
      Whs[((size_t)((be * 32 + chunk) * 4 + nt) * 32 + hic * 16 + m) * 16 + idq] =
          (_Float16)c[r];
    }
  }

  // s1/s2 mat-vec: lanes 0-15 do s1 rows, lanes 16-31 do s2 rows (float4)
  const float* uv = (hi == 0 ? u1 : u2) + e * F_;
  const float* xr = X + ((size_t)(b * N_) + i0 + m) * F_;
  float acc = 0.f;
#pragma unroll
  for (int i = 0; i < 16; ++i) {
    float4 xq = *reinterpret_cast<const float4*>(xr + 4 * i);
    float4 uq = *reinterpret_cast<const float4*>(uv + 4 * i);
    acc += xq.x * uq.x + xq.y * uq.y + xq.z * uq.z + xq.w * uq.w;
  }
  float* dst = (hi == 0 ? s1 : s2);
  dst[(size_t)be * N_ + i0 + m] = acc;
}

// ---------------------------------------------------------------------------
// K2: single-pass flash-style masked softmax + aggregation.
// Per chunk: scores -> online max update -> rescale C -> WMMA accumulate.
// A is streamed exactly once (int4 loads); Wh fragments are single 32B loads.
// ---------------------------------------------------------------------------
__global__ void k_attn(const int* __restrict__ A, const _Float16* __restrict__ Whs,
                       const float* __restrict__ s1g, const float* __restrict__ s2g,
                       _Float16* __restrict__ Hcat) {
  const int lane = threadIdx.x & 31;
  const int wave = threadIdx.x >> 5;
  const int tile = blockIdx.x * 4 + wave;        // 3072 tiles
  const int i0   = (tile & 63) * 16;
  const int e    = (tile >> 6) % E_;
  const int b    = tile / (64 * E_);
  const int m    = lane & 15;
  const int hi   = lane >> 4;
  const int be   = b * E_ + e;

  const int    row  = i0 + m;
  const int*   Arow = A + ((size_t)be * N_ + row) * N_;
  const float* s2p  = s2g + (size_t)be * N_;
  const float  s1v  = s1g[(size_t)be * N_ + row];

  float mx = -3.0e38f, sum = 0.f;
  v8f c[4] = {};

  for (int j0 = 0; j0 < N_; j0 += 32) {
    const int jb = j0 + hi * 8;
    __builtin_prefetch(Arow + jb + 256, 0, 0);
    int4   a0 = *reinterpret_cast<const int4*>(Arow + jb);
    int4   a1 = *reinterpret_cast<const int4*>(Arow + jb + 4);
    int4   a2 = *reinterpret_cast<const int4*>(Arow + jb + 16);
    int4   a3 = *reinterpret_cast<const int4*>(Arow + jb + 20);
    float4 s0 = *reinterpret_cast<const float4*>(s2p + jb);
    float4 s1q = *reinterpret_cast<const float4*>(s2p + jb + 4);
    float4 s2q = *reinterpret_cast<const float4*>(s2p + jb + 16);
    float4 s3q = *reinterpret_cast<const float4*>(s2p + jb + 20);

    int   aa[16] = {a0.x, a0.y, a0.z, a0.w, a1.x, a1.y, a1.z, a1.w,
                    a2.x, a2.y, a2.z, a2.w, a3.x, a3.y, a3.z, a3.w};
    float ss[16] = {s0.x, s0.y, s0.z, s0.w, s1q.x, s1q.y, s1q.z, s1q.w,
                    s2q.x, s2q.y, s2q.z, s2q.w, s3q.x, s3q.y, s3q.z, s3q.w};

    // scores + local max
    float sc[16];
    float cmax = -3.0e38f;
#pragma unroll
    for (int idx = 0; idx < 16; ++idx) {
      float t = s1v + ss[idx];
      t = (t > 0.f) ? t : NEG_SLOPE * t;
      float s = (aa[idx] > 0) ? t : MASK_VAL;
      sc[idx] = s;
      cmax = fmaxf(cmax, s);
    }
    cmax = fmaxf(cmax, __shfl_xor(cmax, 16, 32));   // pair-combine (row m)
    float newm  = fmaxf(mx, cmax);
    float scale = __expf(mx - newm);
    mx = newm;
    sum *= scale;

    // probability fragment
    V16 pa;
#pragma unroll
    for (int idx = 0; idx < 16; ++idx) {
      float p = __expf(sc[idx] - mx);
      sum += p;
      pa.e[idx] = (_Float16)p;
    }

    // rescale accumulators (C row r+8*hi owned stats live on lane r+8*hi)
#pragma unroll
    for (int r = 0; r < 8; ++r) {
      float fr = __shfl(scale, r + 8 * hi, 32);
#pragma unroll
      for (int nt = 0; nt < 4; ++nt) c[nt][r] *= fr;
    }

    // accumulate P @ Wh
#pragma unroll
    for (int nt = 0; nt < 4; ++nt) {
      v16h bw = *reinterpret_cast<const v16h*>(
          Whs + ((size_t)((be * 32 + (j0 >> 5)) * 4 + nt) * 32 + lane) * 16);
      c[nt] = WMMA_F32_F16(pa.v, bw, c[nt]);
    }
  }

  float tot = sum + __shfl_xor(sum, 16, 32);
  float inv = (tot > 0.f) ? 1.f / tot : 0.f;

  _Float16* Hp = Hcat + ((size_t)b * N_ + i0) * CH_ + e * G_;
#pragma unroll
  for (int r = 0; r < 8; ++r) {
    float ir = __shfl(inv, r + 8 * hi, 32);
#pragma unroll
    for (int nt = 0; nt < 4; ++nt)
      Hp[(size_t)(r + 8 * hi) * CH_ + nt * 16 + m] = (_Float16)(c[nt][r] * ir);
  }
}

// ---------------------------------------------------------------------------
// K3: fused MLP 192 -> 256 (ELU) -> 256 (ELU) -> 128 per 16-row tile.
// A-fragments: two 16B vector loads (global or LDS). B-fragments: one 32B
// load from the pre-swizzled f16 weights. LDS bounce = layout transpose.
// ---------------------------------------------------------------------------
__global__ void k_mlp(const _Float16* __restrict__ Hcat,
                      const _Float16* __restrict__ W0z, const float* __restrict__ b0,
                      const _Float16* __restrict__ W1z, const float* __restrict__ b1,
                      const _Float16* __restrict__ W2z, const float* __restrict__ b2,
                      float* __restrict__ out) {
  const int lane = threadIdx.x & 31;
  const int wave = threadIdx.x >> 5;             // 0..1
  const int tile = blockIdx.x * 2 + wave;        // 1024 tiles
  const int r0   = tile * 16;
  const int m    = lane & 15;
  const int hi   = lane >> 4;

  __shared__ _Float16 smem[2][2][16 * H1_];      // 32 KB
  _Float16* act0 = &smem[wave][0][0];
  _Float16* act1 = &smem[wave][1][0];

  // ---- layer 0: Hcat(16x192) @ W0(192x256) + b0, ELU -> act0
  for (int nt = 0; nt < 16; ++nt) {
    v8f c = {};
#pragma unroll
    for (int ks = 0; ks < 6; ++ks) {
      V16 af;
      const _Float16* p0 = Hcat + (size_t)(r0 + m) * CH_ + ks * 32 + 8 * hi;
      af.h8[0] = *reinterpret_cast<const v8h*>(p0);
      af.h8[1] = *reinterpret_cast<const v8h*>(p0 + 16);
      v16h bf = *reinterpret_cast<const v16h*>(
          W0z + ((size_t)(ks * 16 + nt) * 32 + lane) * 16);
      c = WMMA_F32_F16(af.v, bf, c);
    }
    float bias = b0[nt * 16 + m];
#pragma unroll
    for (int r = 0; r < 8; ++r) {
      float x = c[r] + bias;
      x = (x > 0.f) ? x : (__expf(x) - 1.f);
      act0[(r + 8 * hi) * H1_ + nt * 16 + m] = (_Float16)x;
    }
  }
  __syncthreads();

  // ---- layer 1: act0(16x256) @ W1(256x256) + b1, ELU -> act1
  for (int nt = 0; nt < 16; ++nt) {
    v8f c = {};
#pragma unroll
    for (int ks = 0; ks < 8; ++ks) {
      V16 af;
      const _Float16* p0 = act0 + m * H1_ + ks * 32 + 8 * hi;
      af.h8[0] = *reinterpret_cast<const v8h*>(p0);
      af.h8[1] = *reinterpret_cast<const v8h*>(p0 + 16);
      v16h bf = *reinterpret_cast<const v16h*>(
          W1z + ((size_t)(ks * 16 + nt) * 32 + lane) * 16);
      c = WMMA_F32_F16(af.v, bf, c);
    }
    float bias = b1[nt * 16 + m];
#pragma unroll
    for (int r = 0; r < 8; ++r) {
      float x = c[r] + bias;
      x = (x > 0.f) ? x : (__expf(x) - 1.f);
      act1[(r + 8 * hi) * H2_ + nt * 16 + m] = (_Float16)x;
    }
  }
  __syncthreads();

  // ---- layer 2: act1(16x256) @ W2(256x128) + b2 -> out (f32)
  for (int nt = 0; nt < 8; ++nt) {
    v8f c = {};
#pragma unroll
    for (int ks = 0; ks < 8; ++ks) {
      V16 af;
      const _Float16* p0 = act1 + m * H2_ + ks * 32 + 8 * hi;
      af.h8[0] = *reinterpret_cast<const v8h*>(p0);
      af.h8[1] = *reinterpret_cast<const v8h*>(p0 + 16);
      v16h bf = *reinterpret_cast<const v16h*>(
          W2z + ((size_t)(ks * 8 + nt) * 32 + lane) * 16);
      c = WMMA_F32_F16(af.v, bf, c);
    }
    float bias = b2[nt * 16 + m];
#pragma unroll
    for (int r = 0; r < 8; ++r)
      out[(size_t)(r0 + r + 8 * hi) * EMB_ + nt * 16 + m] = c[r] + bias;
  }
}

// ---------------------------------------------------------------------------
extern "C" void kernel_launch(void* const* d_in, const int* in_sizes, int n_in,
                              void* d_out, int out_size, void* d_ws, size_t ws_size,
                              hipStream_t stream) {
  const int*   A  = (const int*)  d_in[0];
  const float* X  = (const float*)d_in[1];
  const float* Ws = (const float*)d_in[2];
  const float* Wq = (const float*)d_in[3];
  const float* Wv = (const float*)d_in[4];
  const float* a1 = (const float*)d_in[5];
  const float* a2 = (const float*)d_in[6];
  const float* W0 = (const float*)d_in[7];
  const float* b0 = (const float*)d_in[8];
  const float* W1 = (const float*)d_in[9];
  const float* b1 = (const float*)d_in[10];
  const float* W2 = (const float*)d_in[11];
  const float* b2 = (const float*)d_in[12];
  float* out = (float*)d_out;

  // workspace layout (~12.7 MB, all offsets 4KB-aligned)
  char* ws = (char*)d_ws;
  float*    u1   = (float*)   (ws + 0);
  float*    u2   = (float*)   (ws + 4096);
  float*    s1   = (float*)   (ws + 8192);                  // 196608 B
  float*    s2   = (float*)   (ws + 204800);                // 196608 B
  _Float16* Wsz  = (_Float16*)(ws + 401408);                // 24576 B
  _Float16* W0z  = (_Float16*)(ws + 425984);                // 98304 B
  _Float16* W1z  = (_Float16*)(ws + 524288);                // 131072 B
  _Float16* W2z  = (_Float16*)(ws + 655360);                // 65536 B
  _Float16* Whs  = (_Float16*)(ws + 720896);                // 6291456 B
  _Float16* Hcat = (_Float16*)(ws + 7012352);               // 6291456 B

  k_uvec<<<dim3(1), dim3(192), 0, stream>>>(Wq, Wv, a1, a2, u1, u2);
  k_wswz<<<dim3(3),  dim3(256), 0, stream>>>(Ws, Wsz, F_,  G_,  E_);
  k_wswz<<<dim3(12), dim3(256), 0, stream>>>(W0, W0z, CH_, H1_, 1);
  k_wswz<<<dim3(16), dim3(256), 0, stream>>>(W1, W1z, H1_, H2_, 1);
  k_wswz<<<dim3(8),  dim3(256), 0, stream>>>(W2, W2z, H2_, EMB_, 1);

  k_proj<<<dim3(768), dim3(128), 0, stream>>>(X, Wsz, u1, u2, Whs, s1, s2);
  k_attn<<<dim3(768), dim3(128), 0, stream>>>(A, Whs, s1, s2, Hcat);
  k_mlp <<<dim3(512), dim3(64),  0, stream>>>(Hcat, W0z, b0, W1z, b1, W2z, b2, out);
}
